// HybridSuperLayer_48808008351933
// MI455X (gfx1250) — matrix operand order repaired and checked by reading, not compiled, not run on hardware
//
#include <hip/hip_runtime.h>
#include <hip/hip_bf16.h>
#include <math.h>

typedef unsigned short u16;
typedef __attribute__((ext_vector_type(8)))  u16    u16x8;
typedef __attribute__((ext_vector_type(16))) __bf16 v16bf;
typedef __attribute__((ext_vector_type(8)))  float  v8f;

union Frag { v16bf bf; u16x8 h[2]; };

// fp32 -> bf16, round-to-nearest-even
static __device__ __forceinline__ u16 f32_to_bf16(float f) {
    unsigned int u = __float_as_uint(f);
    u = u + 0x7FFFu + ((u >> 16) & 1u);
    return (u16)(u >> 16);
}

// 32-byte global -> LDS async copy (2x b128), tracked by ASYNCcnt.
static __device__ __forceinline__ void async_copy_32B(unsigned lds_addr, const void* gptr) {
    asm volatile(
        "global_load_async_to_lds_b128 %0, %1, off\n\t"
        "global_load_async_to_lds_b128 %0, %1, off offset:16"
        :
        : "v"(lds_addr), "v"(gptr)
        : "memory");
}
static __device__ __forceinline__ void wait_async0() {
    asm volatile("s_wait_asynccnt 0x0" ::: "memory");
}
static __device__ __forceinline__ unsigned lds_off(const void* p) {
    return (unsigned)(unsigned long long)p;   // low 32 bits of LDS flat addr = LDS byte address
}

// ---------- prep kernels ----------
__global__ void cvt_f32_to_bf16_k(const float* __restrict__ in, u16* __restrict__ out, int n) {
    int i = blockIdx.x * blockDim.x + threadIdx.x;
    int stride = gridDim.x * blockDim.x;
    for (; i < n; i += stride) out[i] = f32_to_bf16(in[i]);
}

// in: U [rows x r] (row=k, col=n), S[r].  out (TRANSPOSED): [r x rows], out[n*rows+k] = bf16(U[k][n]*S[n])
__global__ void scale_cvt_bf16_T_k(const float* __restrict__ U, const float* __restrict__ S,
                                   u16* __restrict__ out, int rows, int r) {
    int i = blockIdx.x * blockDim.x + threadIdx.x;   // coalesced over U
    int stride = gridDim.x * blockDim.x;
    int n_tot = rows * r;
    for (; i < n_tot; i += stride) {
        int k = i / r, n = i - k * r;
        out[(size_t)n * rows + k] = f32_to_bf16(U[i] * S[n]);
    }
}

// ---------- WMMA GEMM ----------
// C[M x N] = A[M x K] (bf16) @ B^T, with B stored [N x K] row-major (K contiguous).
// EPI==0: store bf16, no bias.  EPI==1: +bias, exact-erf GELU, store bf16.  EPI==2: +bias, store fp32.
template <int EPI>
__launch_bounds__(256, 2)
__global__ void gemm_bf16_wmma(const u16* __restrict__ A, const u16* __restrict__ Bm,
                               const float* __restrict__ bias, void* __restrict__ Cout,
                               int M, int N, int K) {
    // Double-buffered LDS tiles; row stride 40 halfwords (80 B) keeps every
    // 16-byte access aligned while skewing banks.
    __shared__ u16 sA[2][128 * 40];
    __shared__ u16 sB[2][128 * 40];

    const int tid  = threadIdx.x;
    const int lane = tid & 31;
    const int wid  = tid >> 5;
    const int wm   = wid & 3;      // 4 waves along M: 32 rows each
    const int wn   = wid >> 2;     // 2 waves along N: 64 cols each
    const int bm   = blockIdx.y * 128;
    const int bn   = blockIdx.x * 128;

    v8f acc[2][4];
#pragma unroll
    for (int i = 0; i < 2; ++i)
#pragma unroll
        for (int j = 0; j < 4; ++j) acc[i][j] = (v8f){0.f,0.f,0.f,0.f,0.f,0.f,0.f,0.f};

    const int row_s = tid >> 1;    // 0..127 : tile row this thread stages
    const int hf    = tid & 1;     // which 16-element half of the 32-wide K tile
    const u16* Ag = A  + (size_t)(bm + row_s) * K + hf * 16;
    const u16* Bg = Bm + (size_t)(bn + row_s) * K + hf * 16;
    const unsigned ldsA0 = lds_off(&sA[0][row_s * 40 + hf * 16]);
    const unsigned ldsB0 = lds_off(&sB[0][row_s * 40 + hf * 16]);
    const unsigned bufBytes = 128 * 40 * 2;

    const int mloc = lane & 15;
    const int kbA  = (lane < 16) ? 0 : 8;    // A frag: lane<16 -> K {0..7,16..23}, else {8..15,24..31}
    const int kbB  = (lane < 16) ? 0 : 8;    // B frag: lane<16 -> K {0..7,16..23}, else {8..15,24..31}

    // prologue: stage tile 0 into buffer 0
    async_copy_32B(ldsA0, Ag);
    async_copy_32B(ldsB0, Bg);
    wait_async0();
    __syncthreads();

    int buf = 0;
    for (int k0 = 0; k0 < K; k0 += 32) {
        // issue async loads for the next k-tile into the other buffer
        if (k0 + 32 < K) {
            const unsigned nb = (buf ^ 1) ? bufBytes : 0u;
            async_copy_32B(ldsA0 + nb, Ag + k0 + 32);
            async_copy_32B(ldsB0 + nb, Bg + k0 + 32);
        }

        // ---- load fragments (ISA 7.12.2 layouts) from current buffer ----
        Frag a[2];
#pragma unroll
        for (int i = 0; i < 2; ++i) {
            const u16* p = &sA[buf][(wm * 32 + i * 16 + mloc) * 40 + kbA];
            a[i].h[0] = *(const u16x8*)p;
            a[i].h[1] = *(const u16x8*)(p + 16);
        }
        Frag b[4];
#pragma unroll
        for (int j = 0; j < 4; ++j) {
            const u16* p = &sB[buf][(wn * 64 + j * 16 + mloc) * 40 + kbB];
            b[j].h[0] = *(const u16x8*)p;
            b[j].h[1] = *(const u16x8*)(p + 16);
        }

        // ---- 8x v_wmma_f32_16x16x32_bf16, fp32 accumulate ----
#pragma unroll
        for (int i = 0; i < 2; ++i)
#pragma unroll
            for (int j = 0; j < 4; ++j)
                acc[i][j] = __builtin_amdgcn_wmma_f32_16x16x32_bf16(
                    false, a[i].bf, false, b[j].bf, (short)0, acc[i][j], false, false);

        // next tile fully in LDS (this wave) + all waves done reading current
        wait_async0();
        __syncthreads();
        buf ^= 1;
    }

    // ---- epilogue: C/D layout: VGPR r -> (M = r + (lane<16?0:8), N = lane&15) ----
    const int n_l   = lane & 15;
    const int m_off = (lane < 16) ? 0 : 8;
#pragma unroll
    for (int i = 0; i < 2; ++i) {
#pragma unroll
        for (int j = 0; j < 4; ++j) {
            const int col = bn + wn * 64 + j * 16 + n_l;
            float bv = 0.0f;
            if (EPI != 0) bv = bias[col];
#pragma unroll
            for (int r = 0; r < 8; ++r) {
                const int row = bm + wm * 32 + i * 16 + m_off + r;
                float v = acc[i][j][r];
                if (EPI == 1) {  // bias + exact GELU (erf)
                    v += bv;
                    v = 0.5f * v * (1.0f + erff(v * 0.70710678118654752f));
                } else if (EPI == 2) {
                    v += bv;
                }
                if (EPI == 2)
                    ((float*)Cout)[(size_t)row * N + col] = v;
                else
                    ((u16*)Cout)[(size_t)row * N + col] = f32_to_bf16(v);
            }
        }
    }
}

// ---------- host ----------
extern "C" void kernel_launch(void* const* d_in, const int* in_sizes, int n_in,
                              void* d_out, int out_size, void* d_ws, size_t ws_size,
                              hipStream_t stream) {
    constexpr int Bb = 4, S = 2048, D = 1024, DFF = 4096, R = 512;
    constexpr int M = Bb * S;                 // 8192 rows

    const float* x       = (const float*)d_in[0];
    const float* fc_U    = (const float*)d_in[1];
    const float* fc_S    = (const float*)d_in[2];
    const float* fc_V    = (const float*)d_in[3];
    const float* fc_b    = (const float*)d_in[4];
    const float* pj_U    = (const float*)d_in[5];
    const float* pj_S    = (const float*)d_in[6];
    const float* pj_V    = (const float*)d_in[7];
    const float* pj_b    = (const float*)d_in[8];
    float*       out     = (float*)d_out;

    char* ws = (char*)d_ws;
    size_t off = 0;
    auto carve = [&](size_t bytes) -> void* {
        void* p = ws + off;
        off += (bytes + 255) & ~(size_t)255;
        return p;
    };
    u16* Xbf  = (u16*)carve((size_t)M * D * 2);     // bf16(x)                     16 MB
    u16* A1t  = (u16*)carve((size_t)D * R * 2);     // bf16(fc_U*fc_S)^T  [R x D]   1 MB
    u16* Vfc  = (u16*)carve((size_t)DFF * R * 2);   // bf16(fc_V)       [DFF x R]   4 MB
    u16* A2t  = (u16*)carve((size_t)DFF * R * 2);   // bf16(pj_U*pj_S)^T [R x DFF]  4 MB
    u16* Vpj  = (u16*)carve((size_t)D * R * 2);     // bf16(pj_V)         [D x R]   1 MB
    u16* T1   = (u16*)carve((size_t)M * R * 2);     // x @ (U1*S1)                  8 MB
    u16* H    = (u16*)carve((size_t)M * DFF * 2);   // gelu(fc out)                64 MB
    u16* T2   = (u16*)carve((size_t)M * R * 2);     // H @ (U2*S2)                  8 MB
    (void)ws_size; (void)in_sizes; (void)n_in; (void)out_size;

    // prep: fp32 -> bf16; fold diagonal S into U and store U-factors transposed
    // so every GEMM consumes B as [N x K] with K contiguous.
    {
        const int thr = 256;
        auto blocks = [&](int n) { int b = (n + thr - 1) / thr; return b > 4096 ? 4096 : b; };
        cvt_f32_to_bf16_k<<<blocks(M * D), thr, 0, stream>>>(x, Xbf, M * D);
        scale_cvt_bf16_T_k<<<blocks(D * R), thr, 0, stream>>>(fc_U, fc_S, A1t, D, R);
        cvt_f32_to_bf16_k<<<blocks(DFF * R), thr, 0, stream>>>(fc_V, Vfc, DFF * R);
        scale_cvt_bf16_T_k<<<blocks(DFF * R), thr, 0, stream>>>(pj_U, pj_S, A2t, DFF, R);
        cvt_f32_to_bf16_k<<<blocks(D * R), thr, 0, stream>>>(pj_V, Vpj, D * R);
    }

    // Low-rank path: never materialize [D x DFF] weights.
    // G1: T1 = Xbf @ A1t^T          [8192x1024] x [512x1024]^T
    gemm_bf16_wmma<0><<<dim3(R / 128, M / 128), 256, 0, stream>>>(
        Xbf, A1t, nullptr, T1, M, R, D);
    // G2: H = gelu(T1 @ Vfc^T + b)  [8192x512]  x [4096x512]^T
    gemm_bf16_wmma<1><<<dim3(DFF / 128, M / 128), 256, 0, stream>>>(
        T1, Vfc, fc_b, H, M, DFF, R);
    // G3: T2 = H @ A2t^T            [8192x4096] x [512x4096]^T
    gemm_bf16_wmma<0><<<dim3(R / 128, M / 128), 256, 0, stream>>>(
        H, A2t, nullptr, T2, M, R, DFF);
    // G4: out = T2 @ Vpj^T + b      [8192x512]  x [1024x512]^T -> fp32
    gemm_bf16_wmma<2><<<dim3(D / 128, M / 128), 256, 0, stream>>>(
        T2, Vpj, pj_b, out, M, D, R);
}
